// YvSelectiveScan_72112500900641
// MI455X (gfx1250) — matrix operand order repaired and checked by reading, not compiled, not run on hardware
//
#include <hip/hip_runtime.h>
#include <hip/hip_bf16.h>
#include <math.h>

// Problem constants (match reference)
constexpr int B_ = 2, T_ = 2048, M_ = 768, R_ = 64, S_ = 16;
constexpr int BT = B_ * T_;              // 4096 rows
constexpr float LOG_EPS = -23.02585093f; // log(1e-10)

typedef __attribute__((ext_vector_type(16))) __bf16 v16bf;
typedef __attribute__((ext_vector_type(8)))  float  v8f;
typedef __attribute__((ext_vector_type(4)))  unsigned int v4u;
typedef __attribute__((ext_vector_type(8)))  int    v8i;
typedef __attribute__((ext_vector_type(4)))  int    v4i;

#if defined(__has_builtin)
#if __has_builtin(__builtin_amdgcn_tensor_load_to_lds)
#define HAVE_TDM 1
#endif
#endif

// ---------------------------------------------------------------------------
// WMMA fragment helpers (wave32, bf16 16x16x32, contiguous k order, same in A and B)
// A 16x32: lane l (0..15) + half hi: row = l, elems e -> K = kk + hi*16 + e
// B 32x16: lane l = column n, elems e -> K = kk + hi*16 + e
// ---------------------------------------------------------------------------
__device__ __forceinline__ v16bf load_a_frag(const float* __restrict__ rowp,
                                             int kk, int hi) {
  const float4* p0 = reinterpret_cast<const float4*>(rowp + kk + hi * 16);
  float4 x0 = p0[0], x1 = p0[1], y0 = p0[2], y1 = p0[3];
  v16bf r;
  r[0]  = (__bf16)x0.x; r[1]  = (__bf16)x0.y; r[2]  = (__bf16)x0.z; r[3]  = (__bf16)x0.w;
  r[4]  = (__bf16)x1.x; r[5]  = (__bf16)x1.y; r[6]  = (__bf16)x1.z; r[7]  = (__bf16)x1.w;
  r[8]  = (__bf16)y0.x; r[9]  = (__bf16)y0.y; r[10] = (__bf16)y0.z; r[11] = (__bf16)y0.w;
  r[12] = (__bf16)y1.x; r[13] = (__bf16)y1.y; r[14] = (__bf16)y1.z; r[15] = (__bf16)y1.w;
  return r;
}

__device__ __forceinline__ v16bf load_b_frag(const float* __restrict__ rowp,
                                             int kk, int hi) {
  const float4* p = reinterpret_cast<const float4*>(rowp + kk + hi * 16);
  float4 x0 = p[0], x1 = p[1], x2 = p[2], x3 = p[3];
  v16bf r;
  r[0]  = (__bf16)x0.x; r[1]  = (__bf16)x0.y; r[2]  = (__bf16)x0.z; r[3]  = (__bf16)x0.w;
  r[4]  = (__bf16)x1.x; r[5]  = (__bf16)x1.y; r[6]  = (__bf16)x1.z; r[7]  = (__bf16)x1.w;
  r[8]  = (__bf16)x2.x; r[9]  = (__bf16)x2.y; r[10] = (__bf16)x2.z; r[11] = (__bf16)x2.w;
  r[12] = (__bf16)x3.x; r[13] = (__bf16)x3.y; r[14] = (__bf16)x3.z; r[15] = (__bf16)x3.w;
  return r;
}

// Fast softplus using hardware transcendentals (v_exp_f32 / v_log_f32).
__device__ __forceinline__ float softplus_fast(float x) {
  float e  = __expf(fminf(x, 20.0f));
  float sp = __logf(1.0f + e);
  return (x > 20.0f) ? x : sp;
}

// ---------------------------------------------------------------------------
// Kernel 1: dt = clamp(softplus(delta @ dt_w^T + dt_b), 1e-6, 10)
//   delta [BT, R], dt_w [M, R] -> dt [BT, M]
//   one 16x16 output tile per wave; K = 64 -> 2 WMMA steps of 32
// ---------------------------------------------------------------------------
__global__ void yv_dtproj_wmma(const float* __restrict__ delta,
                               const float* __restrict__ dt_w,
                               const float* __restrict__ dt_b,
                               float* __restrict__ dt_out) {
  const int lane = threadIdx.x & 31;
  const int wave = threadIdx.x >> 5;
  const int tile = blockIdx.x * 8 + wave;   // 12288 tiles total
  const int nct  = M_ / 16;                 // 48 col tiles
  const int rt   = tile / nct;              // 0..255
  const int ct   = tile % nct;
  const int l    = lane & 15;
  const int hi   = lane >> 4;
  const int row0 = rt * 16;                 // bt base
  const int m    = ct * 16 + l;             // this lane's column

  v8f acc = {};
  const float* arow = delta + (size_t)(row0 + l) * R_;
  const float* brow = dt_w  + (size_t)m * R_;
#pragma unroll
  for (int kk = 0; kk < R_; kk += 32) {
    v16bf a = load_a_frag(arow, kk, hi);
    v16bf b = load_b_frag(brow, kk, hi);
    acc = __builtin_amdgcn_wmma_f32_16x16x32_bf16(
        false, a, false, b, (short)0, acc, false, false);
  }

  const float bias = dt_b[m];
#pragma unroll
  for (int v = 0; v < 8; ++v) {
    int r = row0 + v + 8 * hi;
    float sp = softplus_fast(acc[v] + bias);
    sp = fminf(fmaxf(sp, 1e-6f), 10.0f);
    dt_out[(size_t)r * M_ + m] = sp;
  }
}

// ---------------------------------------------------------------------------
// Kernel 2: Bm = u @ B_w^T, Cm = u @ C_w^T   (u [BT, M], B_w/C_w [S, M])
//   N = S = 16 -> exactly one WMMA tile wide; share A-fragment across both.
//   K = 768 -> 24 WMMA steps of 32. One 16-row tile per wave (256 tiles).
// ---------------------------------------------------------------------------
__global__ void yv_bcproj_wmma(const float* __restrict__ u,
                               const float* __restrict__ B_w,
                               const float* __restrict__ C_w,
                               float* __restrict__ Bm,
                               float* __restrict__ Cm) {
  const int lane = threadIdx.x & 31;
  const int wave = threadIdx.x >> 5;
  const int rt   = blockIdx.x * 8 + wave;   // 0..255
  const int l    = lane & 15;               // column = s index
  const int hi   = lane >> 4;
  const int row0 = rt * 16;

  v8f accB = {};
  v8f accC = {};
  const float* arow  = u   + (size_t)(row0 + l) * M_;
  const float* browB = B_w + (size_t)l * M_;
  const float* browC = C_w + (size_t)l * M_;
#pragma unroll 4
  for (int kk = 0; kk < M_; kk += 32) {
    v16bf a  = load_a_frag(arow, kk, hi);
    v16bf bb = load_b_frag(browB, kk, hi);
    v16bf bc = load_b_frag(browC, kk, hi);
    accB = __builtin_amdgcn_wmma_f32_16x16x32_bf16(
        false, a, false, bb, (short)0, accB, false, false);
    accC = __builtin_amdgcn_wmma_f32_16x16x32_bf16(
        false, a, false, bc, (short)0, accC, false, false);
  }

#pragma unroll
  for (int v = 0; v < 8; ++v) {
    int r = row0 + v + 8 * hi;
    Bm[(size_t)r * S_ + l] = accB[v];
    Cm[(size_t)r * S_ + l] = accC[v];
  }
}

// ---------------------------------------------------------------------------
// TDM: 1-D tensor_load_to_lds of `nelem` f32 (D# per ISA 08_async_tensor §8)
//   group0: count=1 | lds_addr | global_addr[56:0] | type=2
//   group1: data_size=4B, tensor_dim0=nelem, tile_dim0=nelem, tile_dim1=0
// ---------------------------------------------------------------------------
#ifdef HAVE_TDM
__device__ __forceinline__ void tdm_load_1d(unsigned lds_off,
                                            const void* gptr,
                                            unsigned nelem) {
  unsigned long long ga = (unsigned long long)(uintptr_t)gptr;
  v4u g0;
  g0[0] = 1u;                                          // count=1, user mode
  g0[1] = lds_off;                                     // lds_addr (bytes)
  g0[2] = (unsigned)(ga & 0xFFFFFFFFull);              // global_addr[31:0]
  g0[3] = (unsigned)((ga >> 32) & 0x01FFFFFFull)       // global_addr[56:32]
        | (2u << 30);                                  // type=2 ("image")
  v8i g1;
  g1[0] = (int)(2u << 16);                             // wg_mask=0, data_size=4B
  g1[1] = (int)((nelem & 0xFFFFu) << 16);              // tensor_dim0[15:0]
  g1[2] = (int)((nelem >> 16) & 0xFFFFu);              // tensor_dim0[31:16]
  g1[3] = (int)((nelem & 0xFFFFu) << 16);              // tile_dim0 = nelem
  g1[4] = 0;                                           // tile_dim1=0, tile_dim2=0
  g1[5] = 0;                                           // tensor_dim0_stride (unused, 1-D)
  g1[6] = 0;
  g1[7] = 0;
  v4i z4 = {0, 0, 0, 0};
#if defined(__clang_major__) && (__clang_major__ >= 23)
  v8i z8 = {0, 0, 0, 0, 0, 0, 0, 0};
  __builtin_amdgcn_tensor_load_to_lds(g0, g1, z4, z4, z8, 0);
#else
  __builtin_amdgcn_tensor_load_to_lds(g0, g1, z4, z4, 0);
#endif
}
#endif

// ---------------------------------------------------------------------------
// Kernel 3: selective scan, fused y-contraction + skip + clip.
//   thread = (b, m); 16 h-states in registers; Bm/Cm staged into LDS in
//   chunks of 64 timesteps (2 x 4 KB) via the Tensor Data Mover (wave 0
//   issues, s_wait_tensorcnt, workgroup barrier). Sequential in t.
// ---------------------------------------------------------------------------
constexpr int CHUNK = 64;

__global__ void yv_scan(const float* __restrict__ u,
                        const float* __restrict__ dt,
                        const float* __restrict__ A_log,
                        const float* __restrict__ Bm,
                        const float* __restrict__ Cm,
                        const float* __restrict__ D,
                        float* __restrict__ out) {
  const int blocks_per_b = M_ / 256;                       // 3
  const int b = blockIdx.x / blocks_per_b;
  const int m = (blockIdx.x % blocks_per_b) * 256 + threadIdx.x;

  // [0 .. 1023] = B chunk, [1024 .. 2047] = C chunk (8 KB)
  __shared__ __align__(16) float sBC[2 * CHUNK * S_];

  // Per-thread constants: A[s][m] clamped, D[m]
  float As[S_];
#pragma unroll
  for (int s = 0; s < S_; ++s)
    As[s] = fminf(fmaxf(-__expf(A_log[(size_t)s * M_ + m]), -10.0f), -1e-6f);
  const float Dm = D[m];

  float h[S_];
#pragma unroll
  for (int s = 0; s < S_; ++s) h[s] = 0.0f;

  const size_t base_bm = (size_t)b * T_ * M_ + m;          // stride M_ in t
  const size_t base_bs = (size_t)b * T_ * S_;              // stride S_ in t

  for (int t0 = 0; t0 < T_; t0 += CHUNK) {
    __syncthreads();  // previous chunk fully consumed before overwrite
#ifdef HAVE_TDM
    if (threadIdx.x < 32) {  // wave 0 issues both DMAs (TDM ignores EXEC)
      unsigned ldsB = (unsigned)(uintptr_t)(void*)&sBC[0];
      unsigned ldsC = (unsigned)(uintptr_t)(void*)&sBC[CHUNK * S_];
      tdm_load_1d(ldsB, Bm + base_bs + (size_t)t0 * S_, CHUNK * S_);
      tdm_load_1d(ldsC, Cm + base_bs + (size_t)t0 * S_, CHUNK * S_);
      __builtin_amdgcn_s_wait_tensorcnt(0);
    }
#else
    for (int i = threadIdx.x; i < CHUNK * S_; i += 256) {
      sBC[i]              = Bm[base_bs + (size_t)t0 * S_ + i];
      sBC[CHUNK * S_ + i] = Cm[base_bs + (size_t)t0 * S_ + i];
    }
#endif
    __syncthreads();

    // Prefetch next chunk of u/dt for this thread's column.
    if (t0 + CHUNK < T_) {
      __builtin_prefetch(&u[base_bm + (size_t)(t0 + CHUNK) * M_], 0, 1);
      __builtin_prefetch(&dt[base_bm + (size_t)(t0 + CHUNK) * M_], 0, 1);
    }

    for (int ti = 0; ti < CHUNK; ++ti) {
      const int t = t0 + ti;
      const float dtv = dt[base_bm + (size_t)t * M_];
      const float uv  = u[base_bm + (size_t)t * M_];
      const float dtu = dtv * uv;
      float y = 0.0f;
#pragma unroll
      for (int s = 0; s < S_; ++s) {
        float a = __expf(fmaxf(dtv * As[s], LOG_EPS));
        h[s] = fmaf(a, h[s], dtu * sBC[ti * S_ + s]);
        y = fmaf(h[s], sBC[CHUNK * S_ + ti * S_ + s], y);
      }
      float o = fmaf(uv, Dm, y);
      o = fminf(fmaxf(o, -1e4f), 1e4f);
      out[base_bm + (size_t)t * M_] = o;
    }
  }
}

// ---------------------------------------------------------------------------
// Launch
// ---------------------------------------------------------------------------
extern "C" void kernel_launch(void* const* d_in, const int* in_sizes, int n_in,
                              void* d_out, int out_size, void* d_ws, size_t ws_size,
                              hipStream_t stream) {
  (void)in_sizes; (void)n_in; (void)out_size; (void)ws_size;
  const float* u     = (const float*)d_in[0];
  const float* delta = (const float*)d_in[1];
  const float* dt_w  = (const float*)d_in[2];
  const float* dt_b  = (const float*)d_in[3];
  const float* A_log = (const float*)d_in[4];
  const float* B_w   = (const float*)d_in[5];
  const float* C_w   = (const float*)d_in[6];
  const float* D     = (const float*)d_in[7];
  float* out = (float*)d_out;

  // Workspace layout: dt [BT*M], Bm [BT*S], Cm [BT*S]  (~13.1 MB total)
  float* dt = (float*)d_ws;
  float* Bm = dt + (size_t)BT * M_;
  float* Cm = Bm + (size_t)BT * S_;

  // 12288 dt-tiles / 8 waves per 256-thread block
  yv_dtproj_wmma<<<(BT / 16) * (M_ / 16) / 8, 256, 0, stream>>>(delta, dt_w, dt_b, dt);
  // 256 row tiles / 8 waves per block
  yv_bcproj_wmma<<<(BT / 16) / 8, 256, 0, stream>>>(u, B_w, C_w, Bm, Cm);
  // B_ * (M_/256) = 6 blocks, 256 threads
  yv_scan<<<B_ * (M_ / 256), 256, 0, stream>>>(u, dt, A_log, Bm, Cm, D, out);
}
